// MixtureOfExperts_8555574854179
// MI455X (gfx1250) — matrix operand order repaired and checked by reading, not compiled, not run on hardware
//
#include <hip/hip_runtime.h>
#include <hip/hip_bf16.h>
#include <math.h>

typedef __attribute__((ext_vector_type(16))) _Float16 v16h;
typedef __attribute__((ext_vector_type(4)))  _Float16 v4h;
typedef __attribute__((ext_vector_type(8)))  float    v8f;

#define T_TOKENS 4096
#define DDIM 1024
#define EDIM 8
#define FDIM 4096
#define BM 128
#define BN 128
#define BK 32
#define LSTR 40   // padded LDS row stride in halves (80B; 8/16B-aligned chunks, conflict-free)

// ---- WMMA fragment loaders (per CDNA5 ISA 7.12.2 layouts) ----
// A 16x32 f16: lane m (0-15): halves 0..7 = K kb..kb+7, halves 8..15 = K kb+16..kb+23, kb=(lane>=16)?8:0
__device__ inline v16h ldfragA(const _Float16* lds, int lane) {
  int row = lane & 15;
  int kb  = (lane >> 4) << 3;
  const _Float16* p = lds + row * LSTR + kb;
  v16h r;
#pragma unroll
  for (int j = 0; j < 8; ++j) { r[j] = p[j]; r[j + 8] = p[j + 16]; }
  return r;
}
// B 32x16 f16 (Bt stored [N][K] in LDS): lane n: halves 0..7 = K kb..kb+7, halves 8..15 = K kb+16..kb+23, kb=(lane>=16)?8:0
__device__ inline v16h ldfragB(const _Float16* lds, int lane) {
  int col = lane & 15;
  int kb  = (lane >> 4) << 3;
  const _Float16* p = lds + col * LSTR + kb;
  v16h r;
#pragma unroll
  for (int j = 0; j < 8; ++j) { r[j] = p[j]; r[j + 8] = p[j + 16]; }
  return r;
}

__device__ inline float gelu_exact(float v) {
  return 0.5f * v * (1.0f + erff(v * 0.70710678118654752f));
}

// Stage a 4k x 4n micro-tile of a row-major f32 matrix, packed for transposed LDS layout.
// bh[dn] holds halves (k..k+3) for column n+dn -> single ds_store_b64 per dn.
__device__ inline void load_pack_b(const float* __restrict__ src, size_t ld,
                                   int krow, int ncol, v4h bh[4]) {
  float tmp[4][4];
#pragma unroll
  for (int dk = 0; dk < 4; ++dk) {
    const float* wp = src + (size_t)(krow + dk) * ld + ncol;
#pragma unroll
    for (int dn = 0; dn < 4; ++dn) tmp[dk][dn] = wp[dn];
  }
#pragma unroll
  for (int dn = 0; dn < 4; ++dn)
#pragma unroll
    for (int dk = 0; dk < 4; ++dk) bh[dn][dk] = (_Float16)tmp[dk][dn];
}

// ---------------- zero / init ----------------
__global__ __launch_bounds__(256) void moe_zero(float* out, int nout,
                                                int* counts, int* counts2, float* probsum) {
  size_t i = (size_t)blockIdx.x * blockDim.x + threadIdx.x;
  size_t stride = (size_t)gridDim.x * blockDim.x;
  for (size_t j = i; j < (size_t)nout; j += stride) out[j] = 0.0f;
  if (blockIdx.x == 0 && threadIdx.x < EDIM) {
    counts[threadIdx.x] = 0;
    counts2[threadIdx.x] = 0;
    probsum[threadIdx.x] = 0.0f;
  }
}

// ---------------- router: one wave32 per token ----------------
__global__ __launch_bounds__(256) void moe_router(
    const float* __restrict__ x, const float* __restrict__ rw, const float* __restrict__ rb,
    int* __restrict__ counts, float* __restrict__ probsum,
    int* __restrict__ e1a, int* __restrict__ e2a,
    float* __restrict__ w1a, float* __restrict__ w2a) {
  __shared__ float sp[EDIM];
  int tid = threadIdx.x, lane = tid & 31, wave = tid >> 5;
  if (tid < EDIM) sp[tid] = 0.0f;
  __syncthreads();

  int t = blockIdx.x * 8 + wave;
  float acc[EDIM];
#pragma unroll
  for (int e = 0; e < EDIM; ++e) acc[e] = 0.0f;
  const float* xp = x + (size_t)t * DDIM;
  for (int k = 0; k < DDIM / 32; ++k) {
    int d = k * 32 + lane;              // coalesced across the wave
    float xv = xp[d];
    const float* r = rw + d * EDIM;
#pragma unroll
    for (int e = 0; e < EDIM; ++e) acc[e] += xv * r[e];
  }
#pragma unroll
  for (int e = 0; e < EDIM; ++e)
    for (int m = 16; m >= 1; m >>= 1) acc[e] += __shfl_xor(acc[e], m, 32);

  if (lane == 0) {
    float l[EDIM], p[EDIM];
    float mx = -1e30f;
#pragma unroll
    for (int e = 0; e < EDIM; ++e) { l[e] = acc[e] + rb[e]; mx = fmaxf(mx, l[e]); }
    float s = 0.0f;
#pragma unroll
    for (int e = 0; e < EDIM; ++e) { p[e] = expf(l[e] - mx); s += p[e]; }
    float inv = 1.0f / s;
#pragma unroll
    for (int e = 0; e < EDIM; ++e) p[e] *= inv;
    // top-2, first index wins ties (matches lax.top_k)
    int i1 = 0; float m1 = p[0];
#pragma unroll
    for (int e = 1; e < EDIM; ++e) if (p[e] > m1) { m1 = p[e]; i1 = e; }
    int i2 = -1; float m2 = -1.0f;
#pragma unroll
    for (int e = 0; e < EDIM; ++e) if (e != i1 && p[e] > m2) { m2 = p[e]; i2 = e; }
    float denom = m1 + m2 + 1e-8f;
    e1a[t] = i1; e2a[t] = i2;
    w1a[t] = m1 / denom; w2a[t] = m2 / denom;
    atomicAdd(&counts[i1], 1);
    atomicAdd(&counts[i2], 1);
#pragma unroll
    for (int e = 0; e < EDIM; ++e) atomicAdd(&sp[e], p[e]);
  }
  __syncthreads();
  if (tid < EDIM) atomicAdd(&probsum[tid], sp[tid]);
}

// ---------------- prefix scan + load-balance loss ----------------
__global__ void moe_prefix(const int* __restrict__ counts, int* __restrict__ offsets,
                           int* __restrict__ counts2, const float* __restrict__ probsum,
                           float* __restrict__ out_loss) {
  if (threadIdx.x == 0 && blockIdx.x == 0) {
    int acc = 0;
    for (int e = 0; e < EDIM; ++e) { offsets[e] = acc; acc += counts[e]; counts2[e] = 0; }
    float avg[EDIM], mean = 0.0f;
    for (int e = 0; e < EDIM; ++e) { avg[e] = probsum[e] / (float)T_TOKENS; mean += avg[e]; }
    mean /= (float)EDIM;
    float var = 0.0f;
    for (int e = 0; e < EDIM; ++e) { float d = avg[e] - mean; var += d * d; }
    var /= (float)(EDIM - 1);   // ddof=1
    *out_loss = 0.01f * var;
  }
}

// ---------------- scatter tokens into compacted per-expert lists ----------------
__global__ __launch_bounds__(256) void moe_assign(
    const int* __restrict__ e1a, const int* __restrict__ e2a,
    const float* __restrict__ w1a, const float* __restrict__ w2a,
    const int* __restrict__ offsets, int* __restrict__ counts2,
    int* __restrict__ list, float* __restrict__ wlist) {
  int t = blockIdx.x * 256 + threadIdx.x;
  if (t >= T_TOKENS) return;
  int e1 = e1a[t], e2 = e2a[t];
  int r1 = atomicAdd(&counts2[e1], 1);
  list[offsets[e1] + r1] = t; wlist[offsets[e1] + r1] = w1a[t];
  int r2 = atomicAdd(&counts2[e2], 1);
  list[offsets[e2] + r2] = t; wlist[offsets[e2] + r2] = w2a[t];
}

// ---------------- GEMM1: H = gelu(gather(x) @ w1[e] + b1[e]), f16 out ----------------
// 128x128x32 tiles; 8 waves, each 32x64 (2x4 WMMA accum); software-pipelined staging.
__global__ __launch_bounds__(256) void moe_gemm1(
    const float* __restrict__ x, const float* __restrict__ w1, const float* __restrict__ b1,
    const int* __restrict__ counts, const int* __restrict__ offsets,
    const int* __restrict__ list, _Float16* __restrict__ H) {
  int e = blockIdx.z;
  int ne = counts[e];
  int rt = blockIdx.x * BM;
  if (rt >= ne) return;
  int n0 = blockIdx.y * BN;
  int off = offsets[e];
  const float* w1e = w1 + (size_t)e * DDIM * FDIM;

  __shared__ _Float16 Al[BM * LSTR];
  __shared__ _Float16 Bl[BN * LSTR];

  int tid = threadIdx.x, lane = tid & 31, wave = tid >> 5;
  int wm = wave & 3, wn = wave >> 2;

  v8f acc[2][4] = {};

  // A staging: 128 rows x 32 k; each thread 16 consecutive halves of one row half
  int ar = tid >> 1, aks = (tid & 1) << 4;
  int arow = min(rt + ar, ne - 1);              // clamp (padding masked on store)
  int tok = list[off + arow];
  const float* xrow = x + (size_t)tok * DDIM + aks;
  // B staging: 4k x 4n micro-tile per thread; lane-consecutive n => coalesced 512B rows
  int bn4 = (tid & 31) << 2;                    // n offset 0..124
  int bk4 = (tid >> 5) << 2;                    // k offset 0..28

  _Float16 ah[16];
  v4h bh[4];
  {
#pragma unroll
    for (int j = 0; j < 16; ++j) ah[j] = (_Float16)xrow[j];
    load_pack_b(w1e, FDIM, bk4, n0 + bn4, bh);
  }

  for (int k0 = 0; k0 < DDIM; k0 += BK) {
    __syncthreads();
#pragma unroll
    for (int j = 0; j < 16; ++j) Al[ar * LSTR + aks + j] = ah[j];
#pragma unroll
    for (int dn = 0; dn < 4; ++dn)
      *(v4h*)(&Bl[(bn4 + dn) * LSTR + bk4]) = bh[dn];   // ds_store_b64
    __syncthreads();
    int k2 = k0 + BK;
    if (k2 < DDIM) {                            // prefetch next tile; overlaps WMMAs below
      const float* xp = xrow + k2;
#pragma unroll
      for (int j = 0; j < 16; ++j) ah[j] = (_Float16)xp[j];
      load_pack_b(w1e, FDIM, k2 + bk4, n0 + bn4, bh);
    }
    v16h a0 = ldfragA(Al + (wm * 32) * LSTR, lane);
    v16h a1 = ldfragA(Al + (wm * 32 + 16) * LSTR, lane);
#pragma unroll
    for (int ni = 0; ni < 4; ++ni) {
      v16h bf = ldfragB(Bl + (wn * 64 + ni * 16) * LSTR, lane);
      acc[0][ni] = __builtin_amdgcn_wmma_f32_16x16x32_f16(false, a0, false, bf, (short)0, acc[0][ni], false, false);
      acc[1][ni] = __builtin_amdgcn_wmma_f32_16x16x32_f16(false, a1, false, bf, (short)0, acc[1][ni], false, false);
    }
  }

  int lc = lane & 15, mr = (lane >> 4) << 3;
#pragma unroll
  for (int ni = 0; ni < 4; ++ni) {
    int col = n0 + wn * 64 + ni * 16 + lc;
    float bias = b1[e * FDIM + col];
#pragma unroll
    for (int mi = 0; mi < 2; ++mi) {
#pragma unroll
      for (int i = 0; i < 8; ++i) {
        int grow = rt + wm * 32 + mi * 16 + mr + i;
        if (grow < ne)
          H[(size_t)(off + grow) * FDIM + col] = (_Float16)gelu_exact(acc[mi][ni][i] + bias);
      }
    }
  }
}

// ---------------- GEMM2: out[t] += w_tok * (H @ w2[e] + b2[e]) ----------------
__global__ __launch_bounds__(256) void moe_gemm2(
    const _Float16* __restrict__ H, const float* __restrict__ w2, const float* __restrict__ b2,
    const int* __restrict__ counts, const int* __restrict__ offsets,
    const int* __restrict__ list, const float* __restrict__ wlist,
    float* __restrict__ out) {
  int e = blockIdx.z;
  int ne = counts[e];
  int rt = blockIdx.x * BM;
  if (rt >= ne) return;
  int n0 = blockIdx.y * BN;
  int off = offsets[e];
  const float* w2e = w2 + (size_t)e * FDIM * DDIM;

  __shared__ _Float16 Al[BM * LSTR];
  __shared__ _Float16 Bl[BN * LSTR];

  int tid = threadIdx.x, lane = tid & 31, wave = tid >> 5;
  int wm = wave & 3, wn = wave >> 2;

  v8f acc[2][4] = {};

  int ar = tid >> 1, aks = (tid & 1) << 4;
  int arow = min(rt + ar, ne - 1);              // clamp (padding masked on store)
  const _Float16* hrow = H + (size_t)(off + arow) * FDIM + aks;
  int bn4 = (tid & 31) << 2;
  int bk4 = (tid >> 5) << 2;

  _Float16 ah[16];
  v4h bh[4];
  {
#pragma unroll
    for (int j = 0; j < 16; ++j) ah[j] = hrow[j];
    load_pack_b(w2e, DDIM, bk4, n0 + bn4, bh);
  }

  for (int k0 = 0; k0 < FDIM; k0 += BK) {
    __syncthreads();
#pragma unroll
    for (int j = 0; j < 16; ++j) Al[ar * LSTR + aks + j] = ah[j];
#pragma unroll
    for (int dn = 0; dn < 4; ++dn)
      *(v4h*)(&Bl[(bn4 + dn) * LSTR + bk4]) = bh[dn];   // ds_store_b64
    __syncthreads();
    int k2 = k0 + BK;
    if (k2 < FDIM) {
      const _Float16* hp = hrow + k2;
#pragma unroll
      for (int j = 0; j < 16; ++j) ah[j] = hp[j];
      load_pack_b(w2e, DDIM, k2 + bk4, n0 + bn4, bh);
    }
    v16h a0 = ldfragA(Al + (wm * 32) * LSTR, lane);
    v16h a1 = ldfragA(Al + (wm * 32 + 16) * LSTR, lane);
#pragma unroll
    for (int ni = 0; ni < 4; ++ni) {
      v16h bf = ldfragB(Bl + (wn * 64 + ni * 16) * LSTR, lane);
      acc[0][ni] = __builtin_amdgcn_wmma_f32_16x16x32_f16(false, a0, false, bf, (short)0, acc[0][ni], false, false);
      acc[1][ni] = __builtin_amdgcn_wmma_f32_16x16x32_f16(false, a1, false, bf, (short)0, acc[1][ni], false, false);
    }
  }

  int lc = lane & 15, mr = (lane >> 4) << 3;
  float bias[4];
#pragma unroll
  for (int ni = 0; ni < 4; ++ni) bias[ni] = b2[e * DDIM + n0 + wn * 64 + ni * 16 + lc];
#pragma unroll
  for (int mi = 0; mi < 2; ++mi) {
#pragma unroll
    for (int i = 0; i < 8; ++i) {
      int grow = rt + wm * 32 + mi * 16 + mr + i;
      if (grow < ne) {
        int t = list[off + grow];
        float wt = wlist[off + grow];
#pragma unroll
        for (int ni = 0; ni < 4; ++ni) {
          int col = n0 + wn * 64 + ni * 16 + lc;
          atomicAdd(&out[(size_t)t * DDIM + col], wt * (acc[mi][ni][i] + bias[ni]));
        }
      }
    }
  }
}

extern "C" void kernel_launch(void* const* d_in, const int* in_sizes, int n_in,
                              void* d_out, int out_size, void* d_ws, size_t ws_size,
                              hipStream_t stream) {
  const float* x  = (const float*)d_in[0];
  const float* rw = (const float*)d_in[1];
  const float* rb = (const float*)d_in[2];
  const float* w1 = (const float*)d_in[3];
  const float* b1 = (const float*)d_in[4];
  const float* w2 = (const float*)d_in[5];
  const float* b2 = (const float*)d_in[6];
  float* out  = (float*)d_out;
  float* loss = out + (size_t)T_TOKENS * DDIM;   // scalar appended after [B,S,D]

  // carve workspace (256B-aligned chunks)
  char* p = (char*)d_ws;
  auto carve = [&](size_t bytes) {
    void* r = (void*)p;
    p += (bytes + 255) & ~(size_t)255;
    return r;
  };
  int*   counts  = (int*)carve(EDIM * sizeof(int));
  int*   counts2 = (int*)carve(EDIM * sizeof(int));
  int*   offsets = (int*)carve(EDIM * sizeof(int));
  float* probsum = (float*)carve(EDIM * sizeof(float));
  int*   e1a     = (int*)carve(T_TOKENS * sizeof(int));
  int*   e2a     = (int*)carve(T_TOKENS * sizeof(int));
  float* w1a     = (float*)carve(T_TOKENS * sizeof(float));
  float* w2a     = (float*)carve(T_TOKENS * sizeof(float));
  int*   list    = (int*)carve(2 * T_TOKENS * sizeof(int));
  float* wlist   = (float*)carve(2 * T_TOKENS * sizeof(float));
  _Float16* H    = (_Float16*)carve((size_t)2 * T_TOKENS * FDIM * sizeof(_Float16)); // 64 MB

  moe_zero<<<4096, 256, 0, stream>>>(out, T_TOKENS * DDIM, counts, counts2, probsum);
  moe_router<<<T_TOKENS / 8, 256, 0, stream>>>(x, rw, rb, counts, probsum, e1a, e2a, w1a, w2a);
  moe_prefix<<<1, 1, 0, stream>>>(counts, offsets, counts2, probsum, loss);
  moe_assign<<<T_TOKENS / 256, 256, 0, stream>>>(e1a, e2a, w1a, w2a, offsets, counts2, list, wlist);
  moe_gemm1<<<dim3(T_TOKENS / BM, FDIM / BN, EDIM), 256, 0, stream>>>(x, w1, b1, counts, offsets, list, H);
  moe_gemm2<<<dim3(T_TOKENS / BM, DDIM / BN, EDIM), 256, 0, stream>>>(H, w2, b2, counts, offsets, list, wlist, out);
}
